// graph_attention_44427141710059
// MI455X (gfx1250) — matrix operand order repaired and checked by reading, not compiled, not run on hardware
//
#include <hip/hip_runtime.h>

#define Bc 2
#define Nc 3072
#define Dc 64
#define Oc 64
#define Hc 4

typedef __attribute__((ext_vector_type(16))) __bf16 v16bf;
typedef __attribute__((ext_vector_type(8)))  __bf16 v8bf;
typedef __attribute__((ext_vector_type(8)))  float  v8f;

#define WMMA_BF16(a, b, c)                                                     \
    __builtin_amdgcn_wmma_f32_16x16x32_bf16(false, (a), false, (b), (short)0,  \
                                            (c), false, false)

__device__ __forceinline__ v16bf concat8(v8bf a, v8bf b) {
    return __builtin_shufflevector(a, b, 0, 1, 2, 3, 4, 5, 6, 7,
                                   8, 9, 10, 11, 12, 13, 14, 15);
}

__device__ __forceinline__ float lane_red_max16(float v) {
    v = fmaxf(v, __shfl_xor(v, 1));
    v = fmaxf(v, __shfl_xor(v, 2));
    v = fmaxf(v, __shfl_xor(v, 4));
    v = fmaxf(v, __shfl_xor(v, 8));
    return v;
}
__device__ __forceinline__ float lane_red_sum16(float v) {
    v += __shfl_xor(v, 1);
    v += __shfl_xor(v, 2);
    v += __shfl_xor(v, 4);
    v += __shfl_xor(v, 8);
    return v;
}

// ---------------------------------------------------------------------------
// One 16x64 output tile of a projection for one wave, fully unrolled.
// TR=false: store row-major [N][O];  TR=true: store transposed [O][N] (for v).
// ---------------------------------------------------------------------------
template <bool TR>
__device__ __forceinline__ void proj_tile(
    const __bf16* __restrict__ sWp,   // W^T [o][d] staged in LDS, bf16
    const float*  __restrict__ bias,  // this head's bias (64 floats, global)
    v16bf a0, v16bf a1,               // x A-fragments (K 0..31 / 32..63)
    size_t bh, int row0w, int lmod, int lhalf,
    __bf16* __restrict__ outp)
{
#pragma unroll
    for (int nt = 0; nt < 4; ++nt) {
        const int ocol = nt * 16 + lmod;
        v16bf b0 = *(const v16bf*)(sWp + ocol * Dc + lhalf * 16);
        v16bf b1 = *(const v16bf*)(sWp + ocol * Dc + 32 + lhalf * 16);
        v8f c = {};
        c = WMMA_BF16(a0, b0, c);
        c = WMMA_BF16(a1, b1, c);
        const float bb = bias[ocol];
#pragma unroll
        for (int r = 0; r < 8; ++r) {
            const int row = row0w + r + 8 * lhalf;
            const __bf16 val = (__bf16)(c[r] + bb);
            if (TR) outp[((bh * Oc) + ocol) * (size_t)Nc + row] = val;
            else    outp[((bh * Nc) + row) * (size_t)Oc + ocol] = val;
        }
    }
}

// ---------------------------------------------------------------------------
// Kernel 1: fused QKV projection.  One block = one (b,h) x 128-row tile.
// ---------------------------------------------------------------------------
__global__ __launch_bounds__(256) void qkv_kernel(
    const float* __restrict__ x,
    const float* __restrict__ Wq, const float* __restrict__ bq,
    const float* __restrict__ Wk, const float* __restrict__ bk,
    const float* __restrict__ Wv, const float* __restrict__ bv,
    __bf16* __restrict__ qout, __bf16* __restrict__ kout,
    __bf16* __restrict__ vtout)
{
    __shared__ __attribute__((aligned(32))) __bf16 sWq[Dc * Oc];
    __shared__ __attribute__((aligned(32))) __bf16 sWk[Dc * Oc];
    __shared__ __attribute__((aligned(32))) __bf16 sWv[Dc * Oc];
    __shared__ __attribute__((aligned(32))) __bf16 sX[128 * Dc];

    const int b    = blockIdx.x / Hc;
    const int h    = blockIdx.x % Hc;
    const int row0 = blockIdx.y * 128;
    const int tid  = threadIdx.x;

    const size_t woff = (size_t)h * Dc * Oc;
    for (int idx = tid; idx < Dc * Oc; idx += 256) {
        const int d = idx / Oc, o = idx % Oc;
        sWq[o * Dc + d] = (__bf16)Wq[woff + idx];   // store W^T in LDS
        sWk[o * Dc + d] = (__bf16)Wk[woff + idx];
        sWv[o * Dc + d] = (__bf16)Wv[woff + idx];
    }
    for (int idx = tid; idx < 128 * Dc; idx += 256) {
        const int r = idx / Dc, d = idx % Dc;
        sX[r * Dc + d] = (__bf16)x[((size_t)b * Nc + row0 + r) * Dc + d];
    }
    __syncthreads();

    const int wave  = tid >> 5;
    const int lane  = tid & 31;
    const int lmod  = lane & 15;
    const int lhalf = lane >> 4;
    const int kb    = lhalf * 16;
    const int rowl  = wave * 16 + lmod;

    // x A-fragments: 16x32 each, contiguous K per lane half (k = 16h + s)
    const __bf16* xr = sX + rowl * Dc;
    const v16bf a0 = concat8(*(const v8bf*)(xr + kb),
                             *(const v8bf*)(xr + 8 + kb));
    const v16bf a1 = concat8(*(const v8bf*)(xr + 32 + kb),
                             *(const v8bf*)(xr + 40 + kb));

    const size_t bh    = (size_t)b * Hc + h;
    const int    row0w = row0 + wave * 16;

    proj_tile<false>(sWq, bq + h * Oc, a0, a1, bh, row0w, lmod, lhalf, qout);
    proj_tile<false>(sWk, bk + h * Oc, a0, a1, bh, row0w, lmod, lhalf, kout);
    proj_tile<true >(sWv, bv + h * Oc, a0, a1, bh, row0w, lmod, lhalf, vtout);
}

// ---------------------------------------------------------------------------
// Kernel 2: flash-style fused attention.  One wave owns 16 query rows of one
// (b,h); streams 32 keys/iter; never materializes the NxN score matrix.
// ---------------------------------------------------------------------------
__global__ __launch_bounds__(128) void attn_kernel(
    const __bf16* __restrict__ q, const __bf16* __restrict__ k,
    const __bf16* __restrict__ vt, const int* __restrict__ edge,
    float* __restrict__ feats)
{
    __shared__ __attribute__((aligned(32))) __bf16 sP[4][16 * 32];

    const int b     = blockIdx.x / Hc;
    const int h     = blockIdx.x % Hc;
    const int wave  = threadIdx.x >> 5;
    const int lane  = threadIdx.x & 31;
    const int lmod  = lane & 15;
    const int lhalf = lane >> 4;
    const int kb    = lhalf * 16;
    const int row0  = blockIdx.y * 64 + wave * 16;
    const size_t bh = (size_t)b * Hc + h;

    // q A-fragments held in registers across the whole key loop
    const __bf16* qrow = q + (bh * Nc + row0 + lmod) * Oc;
    const v16bf aq0 = concat8(*(const v8bf*)(qrow + kb),
                              *(const v8bf*)(qrow + 8 + kb));
    const v16bf aq1 = concat8(*(const v8bf*)(qrow + 32 + kb),
                              *(const v8bf*)(qrow + 40 + kb));

    float m_i[8], l_i[8];
    v8f oacc[4];
#pragma unroll
    for (int r = 0; r < 8; ++r) { m_i[r] = -3.0e38f; l_i[r] = 0.0f; }
#pragma unroll
    for (int nt = 0; nt < 4; ++nt) oacc[nt] = (v8f){};

    __bf16* myP = sP[wave];
    const int*    erow = edge + ((size_t)b * Nc + row0 + lmod) * Nc;  // prefetch
    const __bf16* kpf  = k + (bh * Nc + lane) * Oc;                   // prefetch

    for (int j = 0; j < Nc; j += 32) {
        // warm next tile's streaming inputs (k rows + edge rows)
        if (j + 32 < Nc) {
            __builtin_prefetch(kpf + (size_t)(j + 32) * Oc, 0, 3);
            __builtin_prefetch(erow + j + 32, 0, 3);
        }
        // ---- S = (q k^T)/8, leaky, mask ------------------------------------
        v8f s[2];
#pragma unroll
        for (int nt = 0; nt < 2; ++nt) {
            const __bf16* krow = k + (bh * Nc + j + nt * 16 + lmod) * Oc;
            const v16bf kb0 = *(const v16bf*)(krow + lhalf * 16);      // d 0..31
            const v16bf kb1 = *(const v16bf*)(krow + 32 + lhalf * 16); // d 32..63
            v8f c = {};
            c = WMMA_BF16(aq0, kb0, c);
            c = WMMA_BF16(aq1, kb1, c);
            s[nt] = c;
        }
#pragma unroll
        for (int nt = 0; nt < 2; ++nt) {
#pragma unroll
            for (int r = 0; r < 8; ++r) {
                const int row = row0 + r + 8 * lhalf;
                float sv = s[nt][r] * 0.125f;           // 1/sqrt(64)
                sv = sv > 0.0f ? sv : 0.2f * sv;        // leaky_relu(0.2)
                const int e =
                    edge[((size_t)b * Nc + row) * Nc + j + nt * 16 + lmod];
                if (e == 0) sv -= 1.0e10f;              // adjacency mask
                s[nt][r] = sv;
            }
        }
        // ---- online softmax ------------------------------------------------
        float p0[8], p1[8];
#pragma unroll
        for (int r = 0; r < 8; ++r) {
            const float tm = lane_red_max16(fmaxf(s[0][r], s[1][r]));
            const float nm = fmaxf(m_i[r], tm);
            const float alpha = __expf(m_i[r] - nm);
            m_i[r] = nm;
            p0[r] = __expf(s[0][r] - nm);
            p1[r] = __expf(s[1][r] - nm);
            const float rs = lane_red_sum16(p0[r] + p1[r]);
            l_i[r] = l_i[r] * alpha + rs;
#pragma unroll
            for (int nt = 0; nt < 4; ++nt) oacc[nt][r] *= alpha;
        }
        // ---- transpose P (C-layout -> A-layout) via wave-private LDS -------
#pragma unroll
        for (int r = 0; r < 8; ++r) {
            const int row = r + 8 * lhalf;
            myP[row * 32 + lmod]      = (__bf16)p0[r];
            myP[row * 32 + 16 + lmod] = (__bf16)p1[r];
        }
        const __bf16* prow = myP + lmod * 32;
        const v16bf pa = concat8(*(const v8bf*)(prow + kb),
                                 *(const v8bf*)(prow + 8 + kb));
        // ---- O += P @ V (v^T layout -> contiguous 32B B-operand loads) -----
#pragma unroll
        for (int nt = 0; nt < 4; ++nt) {
            const v16bf vb = *(const v16bf*)(
                vt + (bh * Oc + nt * 16 + lmod) * Nc + j + lhalf * 16);
            oacc[nt] = WMMA_BF16(pa, vb, oacc[nt]);
        }
    }

    // ---- normalize and emit per-head features ------------------------------
#pragma unroll
    for (int r = 0; r < 8; ++r) {
        const int row = row0 + r + 8 * lhalf;
        const float inv = 1.0f / l_i[r];
#pragma unroll
        for (int nt = 0; nt < 4; ++nt) {
            feats[(bh * Nc + row) * Oc + nt * 16 + lmod] = oacc[nt][r] * inv;
        }
    }
}

// ---------------------------------------------------------------------------
// Kernel 3: head mean + final leaky relu.
// ---------------------------------------------------------------------------
__global__ __launch_bounds__(256) void mean_kernel(
    const float* __restrict__ feats, float* __restrict__ out)
{
    const int idx = blockIdx.x * 256 + threadIdx.x;     // over B*N*O
    if (idx >= Bc * Nc * Oc) return;
    const int o = idx % Oc;
    const int n = (idx / Oc) % Nc;
    const int b = idx / (Nc * Oc);
    float s = 0.0f;
#pragma unroll
    for (int h = 0; h < Hc; ++h)
        s += feats[(((size_t)b * Hc + h) * Nc + n) * Oc + o];
    s *= 0.25f;
    out[idx] = s > 0.0f ? s : 0.2f * s;
}

extern "C" void kernel_launch(void* const* d_in, const int* in_sizes, int n_in,
                              void* d_out, int out_size, void* d_ws, size_t ws_size,
                              hipStream_t stream) {
    // setup_inputs order: x, edge_index, Wv, bv, Wq, bq, Wk, bk
    const float* x    = (const float*)d_in[0];
    const int*   edge = (const int*)  d_in[1];
    const float* Wv   = (const float*)d_in[2];
    const float* bv   = (const float*)d_in[3];
    const float* Wq   = (const float*)d_in[4];
    const float* bq   = (const float*)d_in[5];
    const float* Wk   = (const float*)d_in[6];
    const float* bk   = (const float*)d_in[7];
    float* out = (float*)d_out;

    char* ws = (char*)d_ws;
    const size_t qkv_bytes = (size_t)Bc * Hc * Nc * Oc * 2; // bf16
    __bf16* qb    = (__bf16*)(ws);
    __bf16* kbuf  = (__bf16*)(ws + qkv_bytes);
    __bf16* vtbuf = (__bf16*)(ws + 2 * qkv_bytes);
    float*  feats = (float*) (ws + 3 * qkv_bytes);

    qkv_kernel<<<dim3(Bc * Hc, Nc / 128), 256, 0, stream>>>(
        x, Wq, bq, Wk, bk, Wv, bv, qb, kbuf, vtbuf);
    attn_kernel<<<dim3(Bc * Hc, Nc / 64), 128, 0, stream>>>(
        qb, kbuf, vtbuf, edge, feats);
    mean_kernel<<<(Bc * Nc * Oc + 255) / 256, 256, 0, stream>>>(feats, out);
}
